// Attention_6107443495625
// MI455X (gfx1250) — compile-verified, matrix operand order checked by reading, not run on hardware
//
#include <hip/hip_runtime.h>

// ---------------------------------------------------------------------------
// Types for CDNA5 WMMA / TDM (gfx1250, wave32)
// ---------------------------------------------------------------------------
typedef __attribute__((ext_vector_type(16))) __bf16 bf16x16;
typedef __attribute__((ext_vector_type(8)))  float  f32x8;
typedef __attribute__((ext_vector_type(4)))  unsigned int u32x4;
typedef __attribute__((ext_vector_type(8)))  int    i32x8;
typedef __attribute__((ext_vector_type(4)))  int    i32x4;

#define SCALE_ATTN 0.125f   // 64^-0.5

#if defined(__HIP_DEVICE_COMPILE__) && defined(__gfx1250__) && \
    __has_builtin(__builtin_amdgcn_tensor_load_to_lds) &&      \
    __has_builtin(__builtin_amdgcn_s_wait_tensorcnt)
#define ATTN_TDM 1
#else
#define ATTN_TDM 0
#endif

__device__ __forceinline__ __bf16 f2bf(float f) {
    // round-to-nearest-even fp32 -> bf16
    unsigned u = __float_as_uint(f);
    unsigned r = (u + 0x7FFFu + ((u >> 16) & 1u)) >> 16;
    unsigned short s = (unsigned short)r;
    union { unsigned short us; __bf16 b; } cv;
    cv.us = s;
    return cv.b;
}

union FragU { bf16x16 v; int4 q[2]; };

// 16x32 (16-bit) A-fragment / pre-transposed B-fragment loader, ISA layout
// (CDNA5 ISA 7.12.2):
//   lanes 0-15 : row = lane,    K = {kb+0..7, kb+16..23}
//   lanes 16-31: row = lane-16, K = {kb+8..15, kb+24..31}
__device__ __forceinline__ bf16x16 load_frag(const __bf16* base, int ld,
                                             int kb, int lane) {
    int half = lane >> 4;
    int r    = lane & 15;
    const __bf16* p = base + (size_t)r * ld + kb + half * 8;
    FragU f;
    f.q[0] = *(const int4*)(p);
    f.q[1] = *(const int4*)(p + 16);
    return f.v;
}

__device__ __forceinline__ f32x8 wmma_bf16(bf16x16 a, bf16x16 b, f32x8 c) {
    return __builtin_amdgcn_wmma_f32_16x16x32_bf16(false, a, false, b,
                                                   (short)0, c, false, false);
}

#if ATTN_TDM
// Issue a 2-D TDM tile load (bf16 elements) per D# layout, CDNA5 ISA.
// All operands are wave-uniform (single-wave workgroup, blockIdx-derived).
// Toolchain exposes the 6-arg builtin: (g0, g1, g2, g3, g_extra, cpol).
__device__ __forceinline__ void tdm_load_2d(unsigned lds_off, const void* gptr,
                                            int tensor_d0, int tensor_d1,
                                            int tile_d0, int tile_d1,
                                            int stride0, int g1d0) {
    unsigned long long ga = (unsigned long long)(uintptr_t)gptr;
    u32x4 g0;
    g0[0] = 1u;                                   // count=1, user mode
    g0[1] = lds_off;                              // LDS byte address
    g0[2] = (unsigned)(ga & 0xFFFFFFFFull);       // global_addr[31:0]
    g0[3] = (unsigned)((ga >> 32) & 0x01FFFFFFull) | 0x80000000u; // type=2
    i32x8 g1;
    g1[0] = g1d0;                                 // data_size/pad config
    g1[1] = (tensor_d0 & 0xFFFF) << 16;           // tensor_dim0[15:0]
    g1[2] = ((unsigned)tensor_d0 >> 16) | ((tensor_d1 & 0xFFFF) << 16);
    g1[3] = ((unsigned)tensor_d1 >> 16) | ((tile_d0 & 0xFFFF) << 16);
    g1[4] = (tile_d1 & 0xFFFF);                   // tile_dim1, tile_dim2=0
    g1[5] = stride0;                              // tensor_dim0_stride[31:0]
    g1[6] = 0;
    g1[7] = 0;
    i32x4 z4 = {0, 0, 0, 0};
    i32x8 z8 = {0, 0, 0, 0, 0, 0, 0, 0};
    __builtin_amdgcn_tensor_load_to_lds(g0, g1, z4, z4, z8, 0);
}
// group1 dword0: data_size=1 (2B) | pad_enable | pad_interval | pad_amount
#define TDM_CFG_K ((1 << 16) | (1 << 20) | (4 << 22) | (3 << 25)) // pad 16B/128B row
#define TDM_CFG_V ((1 << 16) | (1 << 20) | (3 << 22) | (3 << 25)) // pad 16B/64B row
#endif

// ---------------------------------------------------------------------------
// Conversion / packing kernels
// ---------------------------------------------------------------------------
__global__ __launch_bounds__(256) void cvt_f32_to_bf16_k(
    const float* __restrict__ in, __bf16* __restrict__ out, int n) {
    int i = (blockIdx.x * blockDim.x + threadIdx.x) * 4;   // n % 1024 == 0
    float4 f = *(const float4*)(in + i);
    out[i + 0] = f2bf(f.x);
    out[i + 1] = f2bf(f.y);
    out[i + 2] = f2bf(f.z);
    out[i + 3] = f2bf(f.w);
    (void)n;
}

// in [rows][cols] fp32 -> out [cols][rows] bf16  (weights become [N][K])
__global__ __launch_bounds__(256) void transpose_to_bf16_k(
    const float* __restrict__ in, __bf16* __restrict__ out,
    int rows, int cols) {
    int i = blockIdx.x * blockDim.x + threadIdx.x;
    if (i < rows * cols) {
        int r = i / cols, c = i - r * cols;
        out[(size_t)c * rows + r] = f2bf(in[i]);
    }
}

// ---------------------------------------------------------------------------
// QKV projection: A[16384,1024] x Wt[3072,1024] + bias, scattered into
//   q[b][h][n][64], k[b][h][n][64], vT[b][h][64][n]   (all bf16)
// One wave = 32x64 tile (8 WMMA accumulators, 1.5 b128 loads per WMMA).
// ---------------------------------------------------------------------------
__global__ __launch_bounds__(256) void qkv_gemm_wmma(
    const __bf16* __restrict__ A, const __bf16* __restrict__ Bt,
    const float* __restrict__ bias,
    __bf16* __restrict__ qOut, __bf16* __restrict__ kOut,
    __bf16* __restrict__ vtOut) {
    const int lane = threadIdx.x & 31;
    const int wave = blockIdx.x * (blockDim.x >> 5) + (threadIdx.x >> 5);
    const int mTile  = wave / 48;   // 0..511  (rows /32)
    const int nStrip = wave % 48;   // 0..47   (cols /64)
    const int half = lane >> 4, c0 = lane & 15;

    const __bf16* aBase0 = A + (size_t)mTile * 32 * 1024;
    const __bf16* aBase1 = aBase0 + (size_t)16 * 1024;
    const __bf16* bBase  = Bt + (size_t)(nStrip * 64) * 1024;
    f32x8 zero = {0.f, 0.f, 0.f, 0.f, 0.f, 0.f, 0.f, 0.f};
    f32x8 acc[2][4];
#pragma unroll
    for (int mt = 0; mt < 2; ++mt)
#pragma unroll
        for (int t = 0; t < 4; ++t) acc[mt][t] = zero;

    for (int kb = 0; kb < 1024; kb += 32) {
        __builtin_prefetch(aBase0 + ((lane & 15) * 1024) + kb + 256, 0, 1);
        bf16x16 a0 = load_frag(aBase0, 1024, kb, lane);
        bf16x16 a1 = load_frag(aBase1, 1024, kb, lane);
#pragma unroll
        for (int t = 0; t < 4; ++t) {
            bf16x16 bf = load_frag(bBase + (size_t)t * 16 * 1024, 1024, kb, lane);
            acc[0][t] = wmma_bf16(a0, bf, acc[0][t]);
            acc[1][t] = wmma_bf16(a1, bf, acc[1][t]);
        }
    }

    // C layout: vgpr v, lane -> (m = v + half*8, n = c0) within each 16x16.
#pragma unroll
    for (int t = 0; t < 4; ++t) {
        int f   = nStrip * 64 + t * 16 + c0;      // feature 0..3071
        float bv = bias[f];
        int sel  = f >> 10;                       // 0=q 1=k 2=v
        int rem  = f & 1023;
        int head = rem >> 6;
        int d    = rem & 63;
#pragma unroll
        for (int mt = 0; mt < 2; ++mt)
#pragma unroll
            for (int v = 0; v < 8; ++v) {
                int m   = mTile * 32 + mt * 16 + v + half * 8;  // global row
                int bi  = m >> 10;
                int row = m & 1023;
                __bf16 val = f2bf(acc[mt][t][v] + bv);
                size_t bh = (size_t)(bi * 16 + head);
                if (sel == 0)
                    qOut[(bh * 1024 + row) * 64 + d] = val;
                else if (sel == 1)
                    kOut[(bh * 1024 + row) * 64 + d] = val;
                else
                    vtOut[(bh * 64 + d) * 1024 + row] = val;    // V transposed
            }
    }
}

// ---------------------------------------------------------------------------
// Flash attention: single-wave workgroup handles a 16-query block vs all keys.
// K (32x64) and V^T (64x32) tiles are double-buffered into LDS by the Tensor
// Data Mover (tensor_load_to_lds + s_wait_tensorcnt); fragments come from LDS
// with ds_load_b128 (TDM pad fields make the reads bank-conflict free).
// Per 32-key iteration: 4 WMMA (scores) + 4 WMMA (P@V), online softmax.
// ---------------------------------------------------------------------------
__global__ __launch_bounds__(32) void flash_attn_wmma(
    const __bf16* __restrict__ Q, const __bf16* __restrict__ K,
    const __bf16* __restrict__ Vt, __bf16* __restrict__ O) {
#if ATTN_TDM
    __shared__ __align__(16) __bf16 sK[2][32 * 72]; // 144B row pitch (padded)
    __shared__ __align__(16) __bf16 sV[2][64 * 40]; // 80B row pitch (padded)
#endif
    __shared__ __align__(16) __bf16 sP[16 * 40];    // P staging, 80B pitch
    const int lane = threadIdx.x;
    const int bh   = blockIdx.x >> 6;               // (b*16+h)
    const int q0   = (blockIdx.x & 63) << 4;        // query block start
    const int half = lane >> 4, c0 = lane & 15;

    const __bf16* qB = Q  + (size_t)bh * 1024 * 64 + (size_t)q0 * 64;
    const __bf16* kB = K  + (size_t)bh * 1024 * 64;
    const __bf16* vB = Vt + (size_t)bh * 64 * 1024;

    bf16x16 qa0 = load_frag(qB, 64, 0, lane);    // d = 0..31
    bf16x16 qa1 = load_frag(qB, 64, 32, lane);   // d = 32..63

    f32x8 zero = {0.f, 0.f, 0.f, 0.f, 0.f, 0.f, 0.f, 0.f};
    f32x8 o0 = zero, o1 = zero, o2 = zero, o3 = zero;
    float rmax[8], rsum[8];
#pragma unroll
    for (int v = 0; v < 8; ++v) { rmax[v] = -1e30f; rsum[v] = 0.f; }

#if ATTN_TDM
    // prologue: stage first K/V tiles into buffer 0
    tdm_load_2d((unsigned)(uintptr_t)&sK[0][0], kB, 64, 1024, 64, 32, 64,
                TDM_CFG_K);
    tdm_load_2d((unsigned)(uintptr_t)&sV[0][0], vB, 1024, 64, 32, 64, 1024,
                TDM_CFG_V);
    int cur = 0;
#endif

    for (int j0 = 0; j0 < 1024; j0 += 32) {
#if ATTN_TDM
        if (j0 + 32 < 1024) {   // prefetch next tiles into the other buffer
            tdm_load_2d((unsigned)(uintptr_t)&sK[cur ^ 1][0],
                        kB + (size_t)(j0 + 32) * 64, 64, 1024, 64, 32, 64,
                        TDM_CFG_K);
            tdm_load_2d((unsigned)(uintptr_t)&sV[cur ^ 1][0],
                        vB + (size_t)(j0 + 32), 1024, 64, 32, 64, 1024,
                        TDM_CFG_V);
            __builtin_amdgcn_s_wait_tensorcnt(2);  // current pair complete
        } else {
            __builtin_amdgcn_s_wait_tensorcnt(0);
        }
        __asm__ volatile("" ::: "memory");
        const __bf16* kT = sK[cur];   // [32 keys][72], ld=72
        const __bf16* vT = sV[cur];   // [64 d][40],   ld=40
        bf16x16 k00 = load_frag(kT,            72, 0,  lane);
        bf16x16 k01 = load_frag(kT,            72, 32, lane);
        bf16x16 k10 = load_frag(kT + 16 * 72,  72, 0,  lane);
        bf16x16 k11 = load_frag(kT + 16 * 72,  72, 32, lane);
#else
        bf16x16 k00 = load_frag(kB + (size_t)j0 * 64, 64, 0, lane);
        bf16x16 k01 = load_frag(kB + (size_t)j0 * 64, 64, 32, lane);
        bf16x16 k10 = load_frag(kB + (size_t)(j0 + 16) * 64, 64, 0, lane);
        bf16x16 k11 = load_frag(kB + (size_t)(j0 + 16) * 64, 64, 32, lane);
#endif
        // ---- scores S = Q K^T for 32 keys (two 16-col C tiles) ----
        f32x8 s0 = wmma_bf16(qa0, k00, zero);
        s0 = wmma_bf16(qa1, k01, s0);
        f32x8 s1 = wmma_bf16(qa0, k10, zero);
        s1 = wmma_bf16(qa1, k11, s1);

        // ---- online softmax; row = v + half*8, cols across 16 lanes ----
#pragma unroll
        for (int v = 0; v < 8; ++v) {
            float a = s0[v] * SCALE_ATTN;
            float b = s1[v] * SCALE_ATTN;
            float t = fmaxf(a, b);
            t = fmaxf(t, __shfl_xor(t, 1));
            t = fmaxf(t, __shfl_xor(t, 2));
            t = fmaxf(t, __shfl_xor(t, 4));
            t = fmaxf(t, __shfl_xor(t, 8));
            float nm   = fmaxf(rmax[v], t);
            float corr = __expf(rmax[v] - nm);
            rmax[v] = nm;
            float p0 = __expf(a - nm);
            float p1 = __expf(b - nm);
            rsum[v] = rsum[v] * corr + p0 + p1;     // per-lane partial sum
            o0[v] *= corr; o1[v] *= corr; o2[v] *= corr; o3[v] *= corr;
            int m = v + half * 8;
            sP[m * 40 + c0]      = f2bf(p0);
            sP[m * 40 + 16 + c0] = f2bf(p1);
        }
        // intra-wave DS ops are in order; reload P as an A-fragment (16x32)
        bf16x16 pa = load_frag(sP, 40, 0, lane);

        // ---- O += P @ V : 4 d-tiles ----
#if ATTN_TDM
        o0 = wmma_bf16(pa, load_frag(vT + 0 * 16 * 40, 40, 0, lane), o0);
        o1 = wmma_bf16(pa, load_frag(vT + 1 * 16 * 40, 40, 0, lane), o1);
        o2 = wmma_bf16(pa, load_frag(vT + 2 * 16 * 40, 40, 0, lane), o2);
        o3 = wmma_bf16(pa, load_frag(vT + 3 * 16 * 40, 40, 0, lane), o3);
        cur ^= 1;
#else
        o0 = wmma_bf16(pa, load_frag(vB + (size_t)0 * 16 * 1024, 1024, j0, lane), o0);
        o1 = wmma_bf16(pa, load_frag(vB + (size_t)1 * 16 * 1024, 1024, j0, lane), o1);
        o2 = wmma_bf16(pa, load_frag(vB + (size_t)2 * 16 * 1024, 1024, j0, lane), o2);
        o3 = wmma_bf16(pa, load_frag(vB + (size_t)3 * 16 * 1024, 1024, j0, lane), o3);
#endif
    }

    // ---- epilogue: normalize by row sums, write [b][n][h*64+d] bf16 ----
    const int bi = bh >> 4, h = bh & 15;
#pragma unroll
    for (int v = 0; v < 8; ++v) {
        float s = rsum[v];
        s += __shfl_xor(s, 1);
        s += __shfl_xor(s, 2);
        s += __shfl_xor(s, 4);
        s += __shfl_xor(s, 8);
        float inv = 1.0f / s;
        int row = q0 + v + half * 8;
        size_t base = ((size_t)bi * 1024 + row) * 1024 + h * 64 + c0;
        O[base +  0] = f2bf(o0[v] * inv);
        O[base + 16] = f2bf(o1[v] * inv);
        O[base + 32] = f2bf(o2[v] * inv);
        O[base + 48] = f2bf(o3[v] * inv);
    }
}

// ---------------------------------------------------------------------------
// Output projection: attn[16384,1024] x WoutT[1024,1024] + bias -> fp32 out
// One wave = 32x64 tile (8 accumulators).
// ---------------------------------------------------------------------------
__global__ __launch_bounds__(256) void out_gemm_wmma(
    const __bf16* __restrict__ A, const __bf16* __restrict__ Bt,
    const float* __restrict__ bias, float* __restrict__ out) {
    const int lane = threadIdx.x & 31;
    const int wave = blockIdx.x * (blockDim.x >> 5) + (threadIdx.x >> 5);
    const int mTile  = wave >> 4;   // 0..511
    const int nStrip = wave & 15;   // 0..15
    const int half = lane >> 4, c0 = lane & 15;

    const __bf16* aBase0 = A + (size_t)mTile * 32 * 1024;
    const __bf16* aBase1 = aBase0 + (size_t)16 * 1024;
    const __bf16* bBase  = Bt + (size_t)(nStrip * 64) * 1024;
    f32x8 zero = {0.f, 0.f, 0.f, 0.f, 0.f, 0.f, 0.f, 0.f};
    f32x8 acc[2][4];
#pragma unroll
    for (int mt = 0; mt < 2; ++mt)
#pragma unroll
        for (int t = 0; t < 4; ++t) acc[mt][t] = zero;

    for (int kb = 0; kb < 1024; kb += 32) {
        __builtin_prefetch(aBase0 + ((lane & 15) * 1024) + kb + 256, 0, 1);
        bf16x16 a0 = load_frag(aBase0, 1024, kb, lane);
        bf16x16 a1 = load_frag(aBase1, 1024, kb, lane);
#pragma unroll
        for (int t = 0; t < 4; ++t) {
            bf16x16 bf = load_frag(bBase + (size_t)t * 16 * 1024, 1024, kb, lane);
            acc[0][t] = wmma_bf16(a0, bf, acc[0][t]);
            acc[1][t] = wmma_bf16(a1, bf, acc[1][t]);
        }
    }

#pragma unroll
    for (int t = 0; t < 4; ++t) {
        int col = nStrip * 64 + t * 16 + c0;
        float bv = bias[col];
#pragma unroll
        for (int mt = 0; mt < 2; ++mt)
#pragma unroll
            for (int v = 0; v < 8; ++v) {
                int m = mTile * 32 + mt * 16 + v + half * 8;
                out[(size_t)m * 1024 + col] = acc[mt][t][v] + bv;
            }
    }
}

// ---------------------------------------------------------------------------
// Host launcher
// ---------------------------------------------------------------------------
extern "C" void kernel_launch(void* const* d_in, const int* in_sizes, int n_in,
                              void* d_out, int out_size, void* d_ws,
                              size_t ws_size, hipStream_t stream) {
    (void)in_sizes; (void)n_in; (void)out_size; (void)ws_size;

    const float* x     = (const float*)d_in[0];   // [16,1024,1024]
    const float* w_qkv = (const float*)d_in[1];   // [1024,3072]
    const float* b_qkv = (const float*)d_in[2];   // [3072]
    const float* w_out = (const float*)d_in[3];   // [1024,1024]
    const float* b_out = (const float*)d_in[4];   // [1024]
    float* out = (float*)d_out;                   // [16,1024,1024]

    // workspace carve-up (bf16 buffers, all sizes multiples of 256B)
    char* w = (char*)d_ws;
    __bf16* xb    = (__bf16*)w; w += (size_t)16 * 1024 * 1024 * 2;    // x bf16
    __bf16* wqkvT = (__bf16*)w; w += (size_t)3072 * 1024 * 2;         // [3072][1024]
    __bf16* woutT = (__bf16*)w; w += (size_t)1024 * 1024 * 2;         // [1024][1024]
    __bf16* qbuf  = (__bf16*)w; w += (size_t)16 * 16 * 1024 * 64 * 2; // q[b][h][n][64]
    __bf16* kbuf  = (__bf16*)w; w += (size_t)16 * 16 * 1024 * 64 * 2; // k[b][h][n][64]
    __bf16* vtbuf = (__bf16*)w; w += (size_t)16 * 16 * 64 * 1024 * 2; // vT[b][h][64][n]
    __bf16* attnb = (__bf16*)w; w += (size_t)16 * 1024 * 1024 * 2;    // attn bf16

    // 1) convert / pack
    cvt_f32_to_bf16_k<<<16384, 256, 0, stream>>>(x, xb, 16 * 1024 * 1024);
    transpose_to_bf16_k<<<12288, 256, 0, stream>>>(w_qkv, wqkvT, 1024, 3072);
    transpose_to_bf16_k<<<4096, 256, 0, stream>>>(w_out, woutT, 1024, 1024);

    // 2) QKV projection (24576 waves, 8 per block, 32x64 tiles)
    qkv_gemm_wmma<<<3072, 256, 0, stream>>>(xb, wqkvT, b_qkv, qbuf, kbuf, vtbuf);

    // 3) flash attention (16384 single-wave workgroups, TDM double-buffered)
    flash_attn_wmma<<<16384, 32, 0, stream>>>(qbuf, kbuf, vtbuf, attnb);

    // 4) output projection (8192 waves, 8 per block, 32x64 tiles)
    out_gemm_wmma<<<1024, 256, 0, stream>>>(attnb, woutT, b_out, out);
}
